// Block_31233002177054
// MI455X (gfx1250) — matrix operand order repaired and checked by reading, not compiled, not run on hardware
//
#include <hip/hip_runtime.h>
#include <hip/hip_bf16.h>

typedef _Float16 h8   __attribute__((ext_vector_type(8)));
typedef _Float16 v16h __attribute__((ext_vector_type(16)));
typedef float    v8f  __attribute__((ext_vector_type(8)));

#define BDIM 2
#define TDIM 2048
#define CDIM 1024
#define HDIM 16
#define DHD  64
#define DFFD 4096
#define MROWS (BDIM * TDIM) /* 4096 */

// ---------------- fragment loaders ----------------
__device__ __forceinline__ v16h hcat(h8 lo, h8 hi) {
  return __builtin_shufflevector(lo, hi, 0,1,2,3,4,5,6,7,8,9,10,11,12,13,14,15);
}

// A-matrix 16x32 f16 fragment. Caller passes pointer at (row = m0 + (lane&15), k0).
// Per ISA: lane<16 holds K 0..7 & 16..23 ; lane>=16 holds K 8..15 & 24..31.
__device__ __forceinline__ v16h ld_afrag(const _Float16* p, int lane) {
  const int koff = (lane & 16) ? 8 : 0;
  h8 lo = *(const h8*)(p + koff);
  h8 hi = *(const h8*)(p + koff + 16);
  return hcat(lo, hi);
}

// B-matrix 32x16 f16 fragment from B^T storage [N][K]. Caller passes pointer at
// (col = n0 + (lane&15), k0). lanes 0-15 hold K 0..7 & 16..23, lanes 16-31 hold K 8..15 & 24..31.
__device__ __forceinline__ v16h ld_bfrag(const _Float16* p, int lane) {
  const int koff = (lane & 16) ? 8 : 0;
  h8 lo = *(const h8*)(p + koff);
  h8 hi = *(const h8*)(p + koff + 16);
  return hcat(lo, hi);
}

__device__ __forceinline__ v8f wmma_f16(v16h a, v16h b, v8f c) {
  return __builtin_amdgcn_wmma_f32_16x16x32_f16(false, a, false, b, (short)0, c, false, false);
}

// ---------------- weight convert + transpose: f32 [K][N] -> f16 [N][K] ----------------
// 32x32 LDS-tiled transpose, coalesced on both sides. K, N multiples of 32.
__global__ __launch_bounds__(256)
void cvt_transpose_kernel(const float* __restrict__ in, _Float16* __restrict__ out,
                          int K, int N) {
  __shared__ _Float16 tile[32][33];
  const size_t base = (size_t)blockIdx.z * K * N;
  const int kb = blockIdx.y * 32, nb = blockIdx.x * 32;
#pragma unroll
  for (int i = threadIdx.y; i < 32; i += 8)
    tile[i][threadIdx.x] = (_Float16)in[base + (size_t)(kb + i) * N + (nb + threadIdx.x)];
  __syncthreads();
#pragma unroll
  for (int i = threadIdx.y; i < 32; i += 8)
    out[base + (size_t)(nb + i) * K + (kb + threadIdx.x)] = tile[threadIdx.x][i];
}

// ---------------- layernorm (ddof=1) -> f32 and f16 copies ----------------
__global__ __launch_bounds__(256)
void ln_kernel(const float* __restrict__ in, const float* __restrict__ gamma,
               const float* __restrict__ beta, float* __restrict__ outF,
               _Float16* __restrict__ outH) {
  const int row = blockIdx.x;
  const float* x = in + (size_t)row * CDIM;
  float s = 0.f, s2 = 0.f;
  for (int i = threadIdx.x; i < CDIM; i += 256) {
    float v = x[i];
    s += v; s2 += v * v;
  }
  __shared__ float sh[256], sh2[256];
  sh[threadIdx.x] = s; sh2[threadIdx.x] = s2;
  __syncthreads();
  for (int st = 128; st > 0; st >>= 1) {
    if (threadIdx.x < st) { sh[threadIdx.x] += sh[threadIdx.x + st]; sh2[threadIdx.x] += sh2[threadIdx.x + st]; }
    __syncthreads();
  }
  const float mean = sh[0] * (1.0f / CDIM);
  const float var = (sh2[0] - sh[0] * sh[0] * (1.0f / CDIM)) * (1.0f / (CDIM - 1));
  const float inv = rsqrtf(var + 1e-5f);
  for (int i = threadIdx.x; i < CDIM; i += 256) {
    float v = (x[i] - mean) * inv * gamma[i] + beta[i];
    outF[(size_t)row * CDIM + i] = v;
    outH[(size_t)row * CDIM + i] = (_Float16)v;
  }
}

// ---------------- generic WMMA GEMM: out = act(A@B^T + bias) [+ resid] ----------------
// A: f16 [M][K]; Bt: f16 [N][K]. Wave tile 32x64 (2 A-frags x 4 B-frags -> 8 WMMA/k-step,
// A reused 4x, B reused 2x). Block: 8 waves as 2(M) x 4(N) -> 64 x 256 block tile.
__global__ __launch_bounds__(256)
void gemm_wmma_kernel(const _Float16* __restrict__ A, const _Float16* __restrict__ Bt,
                      const float* __restrict__ bias, const float* __restrict__ resid,
                      float* __restrict__ outF, _Float16* __restrict__ outH,
                      int M, int N, int K, int gelu) {
  const int lane = threadIdx.x & 31;
  const int wave = threadIdx.x >> 5;
  const int m0 = blockIdx.y * 64 + (wave >> 2) * 32;
  const int n0 = blockIdx.x * 256 + (wave & 3) * 64;
  if (m0 >= M || n0 >= N) return;
  const _Float16* pa0 = A + (size_t)(m0 + (lane & 15)) * K;
  const _Float16* pa1 = pa0 + (size_t)16 * K;
  const _Float16* pb = Bt + (size_t)(n0 + (lane & 15)) * K;

  v8f acc[2][4] = {};
#pragma unroll 2
  for (int k0 = 0; k0 < K; k0 += 32) {
    const v16h a0 = ld_afrag(pa0 + k0, lane);
    const v16h a1 = ld_afrag(pa1 + k0, lane);
#pragma unroll
    for (int j = 0; j < 4; ++j) {
      const v16h b = ld_bfrag(pb + (size_t)(j * 16) * K + k0, lane);
      acc[0][j] = wmma_f16(a0, b, acc[0][j]);
      acc[1][j] = wmma_f16(a1, b, acc[1][j]);
    }
  }

  const int hi8 = (lane & 16) ? 8 : 0;
#pragma unroll
  for (int i = 0; i < 2; ++i) {
#pragma unroll
    for (int j = 0; j < 4; ++j) {
      const int nc = n0 + j * 16 + (lane & 15);
      const float bn = bias ? bias[nc] : 0.0f;
#pragma unroll
      for (int r = 0; r < 8; ++r) {
        const size_t mr = (size_t)(m0 + i * 16 + r + hi8);
        float v = acc[i][j][r] + bn;
        if (gelu) v = 0.5f * v * (1.0f + erff(v * 0.70710678118654752f));
        if (resid) v += resid[mr * N + nc];
        if (outF) outF[mr * N + nc] = v;
        if (outH) outH[mr * N + nc] = (_Float16)v;
      }
    }
  }
}

// ---------------- per-head Q/K/V projections; V written transposed [B,H,Dh,T] ----------------
// Wave tile 32(M) x 64(=Dh): 2 A-frags x 4 B-frags -> 8 WMMA/k-step.
__global__ __launch_bounds__(256)
void qkv_wmma_kernel(const _Float16* __restrict__ Xh,
                     const _Float16* __restrict__ Wqt, const _Float16* __restrict__ Wkt,
                     const _Float16* __restrict__ Wvt,
                     const float* __restrict__ bq, const float* __restrict__ bk,
                     const float* __restrict__ bv,
                     _Float16* __restrict__ qo, _Float16* __restrict__ ko,
                     _Float16* __restrict__ vTo) {
  const int lane = threadIdx.x & 31;
  const int wave = threadIdx.x >> 5;
  const int h = blockIdx.z & (HDIM - 1);
  const int which = blockIdx.z >> 4;  // 0=Q 1=K 2=V
  const int m0 = blockIdx.x * 256 + wave * 32;

  const _Float16* Wt; const float* bias;
  if (which == 0)      { Wt = Wqt; bias = bq; }
  else if (which == 1) { Wt = Wkt; bias = bk; }
  else                 { Wt = Wvt; bias = bv; }
  Wt += (size_t)h * DHD * CDIM;

  const _Float16* pa0 = Xh + (size_t)(m0 + (lane & 15)) * CDIM;
  const _Float16* pa1 = pa0 + (size_t)16 * CDIM;
  const _Float16* pb = Wt + (size_t)(lane & 15) * CDIM;

  v8f acc[2][4] = {};
#pragma unroll 2
  for (int k0 = 0; k0 < CDIM; k0 += 32) {
    const v16h a0 = ld_afrag(pa0 + k0, lane);
    const v16h a1 = ld_afrag(pa1 + k0, lane);
#pragma unroll
    for (int j = 0; j < 4; ++j) {
      const v16h b = ld_bfrag(pb + (size_t)(j * 16) * CDIM + k0, lane);
      acc[0][j] = wmma_f16(a0, b, acc[0][j]);
      acc[1][j] = wmma_f16(a1, b, acc[1][j]);
    }
  }

  const int hi8 = (lane & 16) ? 8 : 0;
#pragma unroll
  for (int i = 0; i < 2; ++i) {
#pragma unroll
    for (int j = 0; j < 4; ++j) {
      const int nc = j * 16 + (lane & 15);
      const float bn = bias[h * DHD + nc];
#pragma unroll
      for (int r = 0; r < 8; ++r) {
        const int m = m0 + i * 16 + r + hi8;
        const int b = m >> 11;           // T = 2048
        const int t = m & (TDIM - 1);
        const float v = acc[i][j][r] + bn;
        if (which == 2)
          vTo[((size_t)(b * HDIM + h) * DHD + nc) * TDIM + t] = (_Float16)v;
        else {
          _Float16* o = (which == 0) ? qo : ko;
          o[((size_t)(b * HDIM + h) * TDIM + t) * DHD + nc] = (_Float16)v;
        }
      }
    }
  }
}

// ---------------- causal flash attention, one 16-row query tile per wave ----------------
__global__ __launch_bounds__(128)
void attn_wmma_kernel(const _Float16* __restrict__ q, const _Float16* __restrict__ k,
                      const _Float16* __restrict__ vT, _Float16* __restrict__ outH) {
  __shared__ _Float16 plds[4][16][32];
  const int lane = threadIdx.x & 31;
  const int wave = threadIdx.x >> 5;
  const int bh = blockIdx.y;
  const int b = bh >> 4;
  const int h = bh & (HDIM - 1);
  const int t0 = (blockIdx.x * 4 + wave) * 16;
  const int lc = lane & 15;
  const int hi8 = (lane & 16) ? 8 : 0;

  const _Float16* Q  = q  + (size_t)bh * TDIM * DHD;
  const _Float16* Km = k  + (size_t)bh * TDIM * DHD;
  const _Float16* Vt = vT + (size_t)bh * DHD * TDIM;

  const v16h qf0 = ld_afrag(Q + (size_t)(t0 + lc) * DHD + 0, lane);
  const v16h qf1 = ld_afrag(Q + (size_t)(t0 + lc) * DHD + 32, lane);

  v8f o0 = {}, o1 = {}, o2 = {}, o3 = {};
  float mrun[8], lrun[8];
#pragma unroll
  for (int r = 0; r < 8; ++r) { mrun[r] = -__builtin_inff(); lrun[r] = 0.0f; }

  const float scale = 0.125f;  // Dh^-0.5
  const int tmax = t0 + 15;
  for (int s0 = 0; s0 <= tmax; s0 += 32) {
    v8f s[2];
#pragma unroll
    for (int nt = 0; nt < 2; ++nt) {
      const _Float16* pb = Km + (size_t)(s0 + nt * 16 + lc) * DHD;
      v8f a = {};
      a = wmma_f16(qf0, ld_bfrag(pb + 0, lane), a);
      a = wmma_f16(qf1, ld_bfrag(pb + 32, lane), a);
      s[nt] = a;
    }
    // scale + causal mask (col sc vs row tr)
#pragma unroll
    for (int nt = 0; nt < 2; ++nt) {
      const int sc = s0 + nt * 16 + lc;
#pragma unroll
      for (int r = 0; r < 8; ++r) {
        const int tr = t0 + r + hi8;
        s[nt][r] = (sc <= tr) ? s[nt][r] * scale : -__builtin_inff();
      }
    }
    // online softmax per row (rows live across 16 lanes of each half-wave)
#pragma unroll
    for (int r = 0; r < 8; ++r) {
      float loc = fmaxf(s[0][r], s[1][r]);
      loc = fmaxf(loc, __shfl_xor(loc, 8, 32));
      loc = fmaxf(loc, __shfl_xor(loc, 4, 32));
      loc = fmaxf(loc, __shfl_xor(loc, 2, 32));
      loc = fmaxf(loc, __shfl_xor(loc, 1, 32));
      const float mnew = fmaxf(mrun[r], loc);
      const float corr = __expf(mrun[r] - mnew);
      const float p0 = __expf(s[0][r] - mnew);
      const float p1 = __expf(s[1][r] - mnew);
      float rs = p0 + p1;
      rs += __shfl_xor(rs, 8, 32);
      rs += __shfl_xor(rs, 4, 32);
      rs += __shfl_xor(rs, 2, 32);
      rs += __shfl_xor(rs, 1, 32);
      lrun[r] = lrun[r] * corr + rs;
      mrun[r] = mnew;
      o0[r] *= corr; o1[r] *= corr; o2[r] *= corr; o3[r] *= corr;
      plds[wave][r + hi8][lc]      = (_Float16)p0;
      plds[wave][r + hi8][lc + 16] = (_Float16)p1;
    }
    // P (16x32) @ V (32x64): re-tile P via LDS C-layout -> A-layout
    const v16h pf = ld_afrag(&plds[wave][lc][0], lane);
    o0 = wmma_f16(pf, ld_bfrag(Vt + (size_t)(0 * 16 + lc) * TDIM + s0, lane), o0);
    o1 = wmma_f16(pf, ld_bfrag(Vt + (size_t)(1 * 16 + lc) * TDIM + s0, lane), o1);
    o2 = wmma_f16(pf, ld_bfrag(Vt + (size_t)(2 * 16 + lc) * TDIM + s0, lane), o2);
    o3 = wmma_f16(pf, ld_bfrag(Vt + (size_t)(3 * 16 + lc) * TDIM + s0, lane), o3);
  }
  const size_t mbase = (size_t)(b * TDIM + t0);
#pragma unroll
  for (int r = 0; r < 8; ++r) {
    const float rl = 1.0f / lrun[r];
    _Float16* o = outH + (mbase + r + hi8) * CDIM + h * DHD;
    o[0 * 16 + lc] = (_Float16)(o0[r] * rl);
    o[1 * 16 + lc] = (_Float16)(o1[r] * rl);
    o[2 * 16 + lc] = (_Float16)(o2[r] * rl);
    o[3 * 16 + lc] = (_Float16)(o3[r] * rl);
  }
}

// ---------------- host orchestration ----------------
extern "C" void kernel_launch(void* const* d_in, const int* in_sizes, int n_in,
                              void* d_out, int out_size, void* d_ws, size_t ws_size,
                              hipStream_t stream) {
  const float* x      = (const float*)d_in[0];
  const float* Wq     = (const float*)d_in[1];
  const float* bq     = (const float*)d_in[2];
  const float* Wk     = (const float*)d_in[3];
  const float* bk     = (const float*)d_in[4];
  const float* Wv     = (const float*)d_in[5];
  const float* bv     = (const float*)d_in[6];
  const float* Wo     = (const float*)d_in[7];
  const float* bo     = (const float*)d_in[8];
  const float* gamma1 = (const float*)d_in[9];
  const float* beta1  = (const float*)d_in[10];
  const float* gamma2 = (const float*)d_in[11];
  const float* beta2  = (const float*)d_in[12];
  const float* W1     = (const float*)d_in[13];
  const float* b1     = (const float*)d_in[14];
  const float* W2     = (const float*)d_in[15];
  const float* b2     = (const float*)d_in[16];

  char* ws = (char*)d_ws;
  size_t off = 0;
  auto alloc = [&](size_t bytes) { char* p = ws + off; off += (bytes + 255) & ~(size_t)255; return p; };

  float*    xn1  = (float*)   alloc((size_t)MROWS * CDIM * 4);
  _Float16* xh1  = (_Float16*)alloc((size_t)MROWS * CDIM * 2);
  _Float16* qb   = (_Float16*)alloc((size_t)MROWS * CDIM * 2);
  _Float16* kb   = (_Float16*)alloc((size_t)MROWS * CDIM * 2);
  _Float16* vTb  = (_Float16*)alloc((size_t)MROWS * CDIM * 2);
  _Float16* hb   = (_Float16*)alloc((size_t)MROWS * CDIM * 2);
  float*    x2   = (float*)   alloc((size_t)MROWS * CDIM * 4);
  float*    x3   = (float*)   alloc((size_t)MROWS * CDIM * 4);
  _Float16* x3h  = (_Float16*)alloc((size_t)MROWS * CDIM * 2);
  _Float16* gb   = (_Float16*)alloc((size_t)MROWS * DFFD * 2);
  _Float16* Wqt  = (_Float16*)alloc((size_t)HDIM * CDIM * DHD * 2);
  _Float16* Wkt  = (_Float16*)alloc((size_t)HDIM * CDIM * DHD * 2);
  _Float16* Wvt  = (_Float16*)alloc((size_t)HDIM * CDIM * DHD * 2);
  _Float16* Wot  = (_Float16*)alloc((size_t)CDIM * CDIM * 2);
  _Float16* W1t  = (_Float16*)alloc((size_t)CDIM * DFFD * 2);
  _Float16* W2t  = (_Float16*)alloc((size_t)DFFD * CDIM * 2);
  (void)ws_size; (void)in_sizes; (void)n_in; (void)out_size;

  // weight convert + transpose (tiled, coalesced)
  {
    dim3 blk(32, 8);
    cvt_transpose_kernel<<<dim3(DHD / 32, CDIM / 32, HDIM), blk, 0, stream>>>(Wq, Wqt, CDIM, DHD);
    cvt_transpose_kernel<<<dim3(DHD / 32, CDIM / 32, HDIM), blk, 0, stream>>>(Wk, Wkt, CDIM, DHD);
    cvt_transpose_kernel<<<dim3(DHD / 32, CDIM / 32, HDIM), blk, 0, stream>>>(Wv, Wvt, CDIM, DHD);
    cvt_transpose_kernel<<<dim3(CDIM / 32, CDIM / 32, 1), blk, 0, stream>>>(Wo, Wot, CDIM, CDIM);
    cvt_transpose_kernel<<<dim3(DFFD / 32, CDIM / 32, 1), blk, 0, stream>>>(W1, W1t, CDIM, DFFD);
    cvt_transpose_kernel<<<dim3(CDIM / 32, DFFD / 32, 1), blk, 0, stream>>>(W2, W2t, DFFD, CDIM);
  }

  // LN1
  ln_kernel<<<MROWS, 256, 0, stream>>>(x, gamma1, beta1, xn1, xh1);

  // QKV projections (grid.z = 3*H), 256 rows per block
  qkv_wmma_kernel<<<dim3(MROWS / 256, 1, 3 * HDIM), 256, 0, stream>>>(
      xh1, Wqt, Wkt, Wvt, bq, bk, bv, qb, kb, vTb);

  // causal flash attention
  attn_wmma_kernel<<<dim3(TDIM / 64, BDIM * HDIM), 128, 0, stream>>>(qb, kb, vTb, hb);

  // output projection + residual(xn1) -> x2
  gemm_wmma_kernel<<<dim3(CDIM / 256, MROWS / 64), 256, 0, stream>>>(
      hb, Wot, bo, xn1, x2, nullptr, MROWS, CDIM, CDIM, 0);

  // LN2 -> x3 (f32 residual) + x3h (f16 GEMM input)
  ln_kernel<<<MROWS, 256, 0, stream>>>(x2, gamma2, beta2, x3, x3h);

  // FFN1 + exact GELU -> g (f16)
  gemm_wmma_kernel<<<dim3(DFFD / 256, MROWS / 64), 256, 0, stream>>>(
      x3h, W1t, b1, nullptr, nullptr, gb, MROWS, DFFD, CDIM, 1);

  // FFN2 + residual(x3) -> d_out (f32)
  gemm_wmma_kernel<<<dim3(CDIM / 256, MROWS / 64), 256, 0, stream>>>(
      gb, W2t, b2, x3, (float*)d_out, nullptr, MROWS, CDIM, DFFD, 0);
}
